// BiasedAxialAttention_37417755083157
// MI455X (gfx1250) — matrix operand order repaired and checked by reading, not compiled, not run on hardware
//
#include <hip/hip_runtime.h>
#include <math.h>

// ---------------------------------------------------------------------------
// BiasedAxialAttention (tied triangle attention), CDNA5 / gfx1250, wave32.
// All GEMM work routed through v_wmma_f32_16x16x32_bf16, register-blocked
// 32x32 tiles in the two dominant einsum GEMMs, V kept in transposed layout
// so every fragment load is b128-contiguous. Wave-uniform tile indices are
// forced scalar via readfirstlane so addressing stays on the SALU.
// ---------------------------------------------------------------------------

typedef __bf16 bf16_t;
typedef __attribute__((ext_vector_type(16))) __bf16 v16bf;
typedef __attribute__((ext_vector_type(8)))  float  v8f;

#define LDIM 384
#define DDIM 128
#define NH   4
#define DHD  32
#define NPOS (LDIM * LDIM)   // 147456

// Wave id as a compiler-provable scalar (it is wave-uniform by construction).
__device__ inline int uwave() {
  return __builtin_amdgcn_readfirstlane((int)(threadIdx.x >> 5));
}

// ---------------- WMMA fragment helpers (contiguous k: K = 16*(lane/16) + e) -----

__device__ inline v8f wmma_bf16(v16bf a, v16bf b, v8f c) {
  // 8 args: (neg_a, A, neg_b, B, c_mod, C, reuse_a, reuse_b)
  return __builtin_amdgcn_wmma_f32_16x16x32_bf16(false, a, false, b,
                                                 (short)0, c, false, false);
}

// A 16x32 (MxK), element (row,k) at base[row*stride + k] (k contiguous).
// Lanes 0-15: row M=lane, K=0..15; lanes 16-31: K=16..31.
__device__ inline v16bf frag_a_rowmajor(const bf16_t* base, int stride) {
  int l   = threadIdx.x & 31;
  int row = l & 15;
  int kh  = l >> 4;
  const bf16_t* p = base + row * stride + kh * 16;
  v16bf a;
#pragma unroll
  for (int e = 0; e < 16; ++e) a[e] = p[e];         // K = kh*16 + e
  return a;
}

// B 32x16 (KxN), element (k,n) at base[n*stride + k] (k contiguous per column).
// Lanes 0-15: col N=lane, K=0..15; lanes 16-31: K=16..31.
__device__ inline v16bf frag_b_kcontig(const bf16_t* base, size_t stride) {
  int l   = threadIdx.x & 31;
  int col = l & 15;
  int kh  = l >> 4;
  const bf16_t* p = base + (size_t)col * stride + kh * 16;
  v16bf b;
#pragma unroll
  for (int e = 0; e < 16; ++e) b[e] = p[e];         // K = kh*16 + e
  return b;
}

// ---------------- Kernel 1: transpose(L,L) + LayerNorm + cast to bf16 ------

__global__ void k_ln_transpose(const float* __restrict__ pair,
                               const float* __restrict__ bias,
                               const float* __restrict__ lpw, const float* __restrict__ lpb,
                               const float* __restrict__ lbw, const float* __restrict__ lbb,
                               bf16_t* __restrict__ p_bf,
                               bf16_t* __restrict__ b_bf) {
  int l    = threadIdx.x & 31;
  int pos  = blockIdx.x * 8 + uwave();   // output row (after transpose)
  int i = pos / LDIM, j = pos % LDIM;
  int in_row = j * LDIM + i;             // p[i,j] = pair[j,i]

  float4 xp = ((const float4*)(pair + (size_t)in_row * DDIM))[l];
  float4 xb = ((const float4*)(bias + (size_t)in_row * DDIM))[l];

  float sp  = xp.x + xp.y + xp.z + xp.w;
  float sp2 = xp.x*xp.x + xp.y*xp.y + xp.z*xp.z + xp.w*xp.w;
  float sb  = xb.x + xb.y + xb.z + xb.w;
  float sb2 = xb.x*xb.x + xb.y*xb.y + xb.z*xb.z + xb.w*xb.w;
#pragma unroll
  for (int m = 16; m; m >>= 1) {
    sp  += __shfl_xor(sp,  m, 32);
    sp2 += __shfl_xor(sp2, m, 32);
    sb  += __shfl_xor(sb,  m, 32);
    sb2 += __shfl_xor(sb2, m, 32);
  }
  float mup = sp * (1.0f / DDIM);
  float rsp = rsqrtf(sp2 * (1.0f / DDIM) - mup * mup + 1e-5f);
  float mub = sb * (1.0f / DDIM);
  float rsb = rsqrtf(sb2 * (1.0f / DDIM) - mub * mub + 1e-5f);

  float vp[4] = {xp.x, xp.y, xp.z, xp.w};
  float vb[4] = {xb.x, xb.y, xb.z, xb.w};
#pragma unroll
  for (int e = 0; e < 4; ++e) {
    int d = l * 4 + e;
    p_bf[pos * DDIM + d] = (bf16_t)((vp[e] - mup) * rsp * lpw[d] + lpb[d]);
    b_bf[pos * DDIM + d] = (bf16_t)((vb[e] - mub) * rsb * lbw[d] + lbb[d]);
  }
}

// ---------------- Kernel 2: weights -> transposed bf16 (scales folded) -----

__global__ void k_convert_weights(const float* __restrict__ Wq, const float* __restrict__ Wk,
                                  const float* __restrict__ Wv, const float* __restrict__ Wg,
                                  const float* __restrict__ Wo,
                                  bf16_t* __restrict__ WqT, bf16_t* __restrict__ WkT,
                                  bf16_t* __restrict__ WvT, bf16_t* __restrict__ WgT,
                                  bf16_t* __restrict__ WoT) {
  int t = blockIdx.x * blockDim.x + threadIdx.x;
  if (t >= 5 * DDIM * DDIM) return;
  int m = t / (DDIM * DDIM), idx = t % (DDIM * DDIM);
  int k = idx / DDIM, n = idx % DDIM;
  const float* W; bf16_t* T; float s;
  if (m == 0)      { W = Wq; T = WqT; s = 0.17677669529663689f; }   // 1/sqrt(32)
  else if (m == 1) { W = Wk; T = WkT; s = 0.05103103630798288f; }   // 1/sqrt(384)
  else if (m == 2) { W = Wv; T = WvT; s = 1.0f; }
  else if (m == 3) { W = Wg; T = WgT; s = 1.0f; }
  else             { W = Wo; T = WoT; s = 1.0f; }
  T[n * DDIM + k] = (bf16_t)(W[k * DDIM + n] * s);    // store K-contiguous
}

// ---------------- Kernel 3: q/k/v/gate projections (WMMA) ------------------
// Block: 256 thr = 8 waves; one 16-row tile of p per block staged in LDS.
// Waves 0-1 -> q, 2-3 -> k, 4-5 -> v (stored TRANSPOSED), 6-7 -> gate.

__global__ void k_proj(const bf16_t* __restrict__ p_bf,
                       const bf16_t* __restrict__ WqT, const bf16_t* __restrict__ WkT,
                       const bf16_t* __restrict__ WvT, const bf16_t* __restrict__ WgT,
                       const float* __restrict__ bg,
                       bf16_t* __restrict__ q_bf, bf16_t* __restrict__ k_bf,
                       bf16_t* __restrict__ vT,  bf16_t* __restrict__ g_bf) {
  __shared__ __align__(16) bf16_t tile[16 * DDIM];
  int row0 = blockIdx.x * 16;
  ((uint4*)tile)[threadIdx.x] = ((const uint4*)(p_bf + (size_t)row0 * DDIM))[threadIdx.x];
  __syncthreads();

  int w   = uwave();                 // scalar wave id -> scalar branches below
  int sel = w >> 1;
  int colbase = (w & 1) * 64;
  const bf16_t* WT = (sel == 0) ? WqT : (sel == 1) ? WkT : (sel == 2) ? WvT : WgT;

  int l = threadIdx.x & 31;
  int col_l = l & 15, rb = (l >> 4) * 8;

  for (int nt = 0; nt < 4; ++nt) {
    int col0 = colbase + nt * 16;
    v8f acc = {};
#pragma unroll
    for (int ks = 0; ks < 4; ++ks) {
      v16bf a = frag_a_rowmajor(tile + ks * 32, DDIM);
      v16bf b = frag_b_kcontig(WT + col0 * DDIM + ks * 32, DDIM);
      acc = wmma_bf16(a, b, acc);
    }
    int col = col0 + col_l;
    if (sel == 0) {
#pragma unroll
      for (int r = 0; r < 8; ++r)
        q_bf[(row0 + rb + r) * DDIM + col] = (bf16_t)acc[r];
    } else if (sel == 1) {
#pragma unroll
      for (int r = 0; r < 8; ++r)
        k_bf[(row0 + rb + r) * DDIM + col] = (bf16_t)acc[r];
    } else if (sel == 2) {
      // transposed store: vT[c][pos]; 8 consecutive positions -> one b128.
      union { uint4 u; bf16_t h[8]; } pk;
#pragma unroll
      for (int r = 0; r < 8; ++r) pk.h[r] = (bf16_t)acc[r];
      *((uint4*)(vT + (size_t)col * NPOS + row0 + rb)) = pk.u;
    } else {
      float bgv = bg[col];
#pragma unroll
      for (int r = 0; r < 8; ++r) {
        float x = acc[r] + bgv;
        // sigmoid via v_exp + v_rcp (no IEEE divide sequence)
        g_bf[(row0 + rb + r) * DDIM + col] =
            (bf16_t)__builtin_amdgcn_rcpf(1.0f + __expf(-x));
      }
    }
  }
}

// ---------------- Kernel 4: bias projection (tiny N=4, plain VALU) ---------

__global__ void k_bproj(const bf16_t* __restrict__ b_bf, const float* __restrict__ Wb,
                        float* __restrict__ bproj) {
  int t = blockIdx.x * blockDim.x + threadIdx.x;
  int pos = t >> 2, h = t & 3;
  const bf16_t* row = b_bf + (size_t)pos * DDIM;
  float acc = 0.0f;
#pragma unroll 8
  for (int c = 0; c < DDIM; ++c) acc += (float)row[c] * Wb[c * NH + h];
  bproj[pos * NH + h] = acc;
}

// ---------------- Kernel 5: tied logits attn[h,i,j] (WMMA, K = 384*32) -----
// One 32x32 output tile per wave (2 A-frags x 2 B-frags -> 4 WMMA / K-step).
// 576 wave-tasks: (h, i-block of 32, j-block of 32).

__global__ void k_logits(const bf16_t* __restrict__ q_bf, const bf16_t* __restrict__ k_bf,
                         const float* __restrict__ bproj, float* __restrict__ attn) {
  int task = blockIdx.x * 8 + uwave();              // scalar task id
  int h    = task / 144;
  int rem  = task % 144;
  int i0   = (rem / 12) * 32;
  int j0   = (rem % 12) * 32;

  v8f acc[2][2] = {};
#pragma unroll 2
  for (int n = 0; n < LDIM; ++n) {
    const bf16_t* qb = q_bf + ((n * LDIM + i0) * DDIM) + h * DHD;
    const bf16_t* kb = k_bf + ((n * LDIM + j0) * DDIM) + h * DHD;
    v16bf a0 = frag_a_rowmajor(qb,             DDIM);
    v16bf a1 = frag_a_rowmajor(qb + 16 * DDIM, DDIM);
    v16bf b0 = frag_b_kcontig (kb,             DDIM);
    v16bf b1 = frag_b_kcontig (kb + 16 * DDIM, DDIM);
    acc[0][0] = wmma_bf16(a0, b0, acc[0][0]);
    acc[0][1] = wmma_bf16(a0, b1, acc[0][1]);
    acc[1][0] = wmma_bf16(a1, b0, acc[1][0]);
    acc[1][1] = wmma_bf16(a1, b1, acc[1][1]);
  }

  int l = threadIdx.x & 31;
  int col = l & 15, rb = (l >> 4) * 8;
#pragma unroll
  for (int ti = 0; ti < 2; ++ti)
#pragma unroll
    for (int tj = 0; tj < 2; ++tj)
#pragma unroll
      for (int r = 0; r < 8; ++r) {
        int i = i0 + ti * 16 + rb + r;
        int j = j0 + tj * 16 + col;
        attn[(h * LDIM + i) * LDIM + j] = acc[ti][tj][r] + bproj[(i * LDIM + j) * NH + h];
      }
}

// ---------------- Kernel 6: softmax over j, cast to bf16 -------------------

__global__ void k_softmax(const float* __restrict__ attn, bf16_t* __restrict__ attn_bf) {
  int l = threadIdx.x & 31;
  int row = blockIdx.x * 8 + uwave();    // row in [0, NH*LDIM)
  const float* src = attn + row * LDIM;
  float vals[12];
  float mx = -1e30f;
#pragma unroll
  for (int e = 0; e < 12; ++e) { vals[e] = src[e * 32 + l]; mx = fmaxf(mx, vals[e]); }
#pragma unroll
  for (int m = 16; m; m >>= 1) mx = fmaxf(mx, __shfl_xor(mx, m, 32));
  float s = 0.0f;
#pragma unroll
  for (int e = 0; e < 12; ++e) { vals[e] = __expf(vals[e] - mx); s += vals[e]; }
#pragma unroll
  for (int m = 16; m; m >>= 1) s += __shfl_xor(s, m, 32);
  float inv = __builtin_amdgcn_rcpf(s);
  bf16_t* dst = attn_bf + row * LDIM;
#pragma unroll
  for (int e = 0; e < 12; ++e) dst[e * 32 + l] = (bf16_t)(vals[e] * inv);
}

// ---------------- Kernel 7: out = attn @ V, fused gate (WMMA) --------------
// out[i,kid,h,d] = sum_j attn[h,i,j] * vT[h*32+d][kid*L+j].
// One wave: 32 i-rows x full 32 d (2x2 tiles); all frag loads b128-contiguous.

__global__ void k_av(const bf16_t* __restrict__ attn_bf, const bf16_t* __restrict__ vT,
                     const bf16_t* __restrict__ g_bf, bf16_t* __restrict__ og) {
  int task = blockIdx.x * 8 + uwave();              // scalar task id
  int h    = task / (LDIM * 12);
  int rem  = task % (LDIM * 12);
  int kidx = rem / 12;
  int i0   = (rem % 12) * 32;

  v8f acc[2][2] = {};
#pragma unroll 2
  for (int js = 0; js < 12; ++js) {
    int j0 = js * 32;
    const bf16_t* ab = attn_bf + (h * LDIM + i0) * LDIM + j0;
    const bf16_t* vb = vT + (size_t)(h * DHD) * NPOS + kidx * LDIM + j0;
    v16bf a0 = frag_a_rowmajor(ab,             LDIM);
    v16bf a1 = frag_a_rowmajor(ab + 16 * LDIM, LDIM);
    v16bf b0 = frag_b_kcontig (vb,                     (size_t)NPOS);
    v16bf b1 = frag_b_kcontig (vb + (size_t)16 * NPOS, (size_t)NPOS);
    acc[0][0] = wmma_bf16(a0, b0, acc[0][0]);
    acc[0][1] = wmma_bf16(a0, b1, acc[0][1]);
    acc[1][0] = wmma_bf16(a1, b0, acc[1][0]);
    acc[1][1] = wmma_bf16(a1, b1, acc[1][1]);
  }

  int l = threadIdx.x & 31;
  int col = l & 15, rb = (l >> 4) * 8;
#pragma unroll
  for (int ti = 0; ti < 2; ++ti)
#pragma unroll
    for (int td = 0; td < 2; ++td)
#pragma unroll
      for (int r = 0; r < 8; ++r) {
        int i = i0 + ti * 16 + rb + r;
        size_t idx = (size_t)(i * LDIM + kidx) * DDIM + h * DHD + td * 16 + col;
        og[idx] = (bf16_t)((float)g_bf[idx] * acc[ti][td][r]);
      }
}

// ---------------- Kernel 8: final GEMM @ Wo + bo, transpose on store -------

__global__ void k_out(const bf16_t* __restrict__ og, const bf16_t* __restrict__ WoT,
                      const float* __restrict__ bo, float* __restrict__ out) {
  __shared__ __align__(16) bf16_t tile[16 * DDIM];
  int row0 = blockIdx.x * 16;
  ((uint4*)tile)[threadIdx.x] = ((const uint4*)(og + (size_t)row0 * DDIM))[threadIdx.x];
  __syncthreads();

  int col0 = uwave() * 16;
  v8f acc = {};
#pragma unroll
  for (int ks = 0; ks < 4; ++ks) {
    v16bf a = frag_a_rowmajor(tile + ks * 32, DDIM);
    v16bf b = frag_b_kcontig(WoT + col0 * DDIM + ks * 32, DDIM);
    acc = wmma_bf16(a, b, acc);
  }

  int l = threadIdx.x & 31;
  int col = col0 + (l & 15), rb = (l >> 4) * 8;
  float bov = bo[col];
#pragma unroll
  for (int r = 0; r < 8; ++r) {
    int pos = row0 + rb + r;              // pos = i*LDIM + kidx
    int i = pos / LDIM, kidx = pos % LDIM;
    out[(kidx * LDIM + i) * DDIM + col] = acc[r] + bov;   // transpose (0,2,1,3)
  }
}

// ---------------------------------------------------------------------------

extern "C" void kernel_launch(void* const* d_in, const int* in_sizes, int n_in,
                              void* d_out, int out_size, void* d_ws, size_t ws_size,
                              hipStream_t stream) {
  const float* pair = (const float*)d_in[0];
  const float* bias = (const float*)d_in[1];
  const float* lpw  = (const float*)d_in[2];
  const float* lpb  = (const float*)d_in[3];
  const float* lbw  = (const float*)d_in[4];
  const float* lbb  = (const float*)d_in[5];
  const float* Wq   = (const float*)d_in[6];
  const float* Wk   = (const float*)d_in[7];
  const float* Wv   = (const float*)d_in[8];
  const float* Wb   = (const float*)d_in[9];
  const float* Wg   = (const float*)d_in[10];
  const float* bg   = (const float*)d_in[11];
  const float* Wo   = (const float*)d_in[12];
  const float* bo   = (const float*)d_in[13];
  float* out        = (float*)d_out;

  char* ws = (char*)d_ws;
  size_t off = 0;
  auto alloc = [&](size_t bytes) -> void* {
    void* p = ws + off;
    off = (off + bytes + 255) & ~(size_t)255;
    return p;
  };

  const size_t act = (size_t)NPOS * DDIM * sizeof(bf16_t);   // 37.75 MB each
  bf16_t* p_bf   = (bf16_t*)alloc(act);
  bf16_t* b_bf   = (bf16_t*)alloc(act);
  bf16_t* q_bf   = (bf16_t*)alloc(act);
  bf16_t* k_bf   = (bf16_t*)alloc(act);
  bf16_t* vT     = (bf16_t*)alloc(act);   // transposed: vT[c][pos]
  bf16_t* g_bf   = (bf16_t*)alloc(act);
  bf16_t* og_bf  = (bf16_t*)alloc(act);
  float*  bproj  = (float*)alloc((size_t)NPOS * NH * sizeof(float));
  float*  attn   = (float*)alloc((size_t)NH * LDIM * LDIM * sizeof(float));
  bf16_t* attnbf = (bf16_t*)alloc((size_t)NH * LDIM * LDIM * sizeof(bf16_t));
  bf16_t* WqT    = (bf16_t*)alloc(DDIM * DDIM * sizeof(bf16_t));
  bf16_t* WkT    = (bf16_t*)alloc(DDIM * DDIM * sizeof(bf16_t));
  bf16_t* WvT    = (bf16_t*)alloc(DDIM * DDIM * sizeof(bf16_t));
  bf16_t* WgT    = (bf16_t*)alloc(DDIM * DDIM * sizeof(bf16_t));
  bf16_t* WoT    = (bf16_t*)alloc(DDIM * DDIM * sizeof(bf16_t));

  dim3 blk(256);

  // 1) LN + transpose (8 waves = 8 positions / block)
  k_ln_transpose<<<dim3(NPOS / 8), blk, 0, stream>>>(pair, bias, lpw, lpb, lbw, lbb,
                                                     p_bf, b_bf);
  // 2) weight conversion (transposed, scales folded)
  k_convert_weights<<<dim3((5 * DDIM * DDIM + 255) / 256), blk, 0, stream>>>(
      Wq, Wk, Wv, Wg, Wo, WqT, WkT, WvT, WgT, WoT);
  // 3) q/k/v/gate projections (v written transposed)
  k_proj<<<dim3(NPOS / 16), blk, 0, stream>>>(p_bf, WqT, WkT, WvT, WgT, bg,
                                              q_bf, k_bf, vT, g_bf);
  // 4) bias projection
  k_bproj<<<dim3(NPOS * NH / 256), blk, 0, stream>>>(b_bf, Wb, bproj);
  // 5) tied logits: 576 32x32 wave-tiles
  k_logits<<<dim3(576 / 8), blk, 0, stream>>>(q_bf, k_bf, bproj, attn);
  // 6) softmax over j
  k_softmax<<<dim3(NH * LDIM / 8), blk, 0, stream>>>(attn, attnbf);
  // 7) attn @ V with fused gating: 4*384*12 32x32 wave-tiles
  k_av<<<dim3(NH * LDIM * 12 / 8), blk, 0, stream>>>(attnbf, vT, g_bf, og_bf);
  // 8) final GEMM + bias + transpose back
  k_out<<<dim3(NPOS / 16), blk, 0, stream>>>(og_bf, WoT, bo, out);
}
